// InversePatchEmbedding_38457137168552
// MI455X (gfx1250) — matrix operand order repaired and checked by reading, not compiled, not run on hardware
//
#include <hip/hip_runtime.h>
#include <hip/hip_bf16.h>

// InversePatchEmbedding: out = voxel_shuffle(x @ W^T + b)
//   x [2, 32768, 768] f32, W [256, 768] f32, b [256] f32 -> out [2,4,128,128,128] f32
// bf16x3-split WMMA GEMM (fp32-faithful). 2-D register blocking: each wave owns
// 4 M-tiles x 2 f-tiles; A fragments resident in regs per K-step (read from LDS
// once, converted once by filler waves), B streamed from LDS with 12x reuse.

typedef __bf16 bf16_t;
typedef __attribute__((ext_vector_type(16))) __bf16 v16bf;
typedef __attribute__((ext_vector_type(8)))  float  v8f;
typedef __attribute__((ext_vector_type(4)))  float  v4f;

#define EMBED   768
#define FOUT    256
#define KSTEPS  24                          // K chunks of 32
#define FTILES  16                          // 16x16 feature tiles
#define STEP_HALVES (2 * FTILES * 32 * 16)  // 16384 bf16 = 32KB of B per K-step
#define STEP_UINT4  (STEP_HALVES / 8)       // 2048 uint4 per K-step

#define MT 4      // M-tiles per wave (A resident in regs)
#define FT 2      // f-tiles per wave (B streamed from LDS)
#define MGROUPS 2 // M-groups per workgroup
#define FGROUPS 8 // F-groups per workgroup  (MGROUPS*FGROUPS = 16 waves)
#define ATILES  (MGROUPS * MT)              // 8 A tiles staged per step
#define WGROWS  (ATILES * 16)               // 128 rows per workgroup

// ---------------------------------------------------------------------------
// Kernel 1: pack W [256 x 768] fp32 into per-K-step contiguous WMMA B-fragment
// blocks, split into bf16 hi/lo planes.
// Layout: wpk[ks][plane][ftile][lane][16]  (one K-step = contiguous 32KB)
// Lane data (contiguous k order, same as A; 32x16 K-major tile):
//   lane holds W[ft*16 + (lane&15)][ks*32 + (lane>>4)*16 + 0..15]
// ---------------------------------------------------------------------------
__global__ __launch_bounds__(256) void pack_w_kernel(
    const float* __restrict__ W, bf16_t* __restrict__ wpk) {
  int t = blockIdx.x * blockDim.x + threadIdx.x;   // (ks, ft, lane)
  if (t >= KSTEPS * FTILES * 32) return;
  int lane = t & 31;
  int ft   = (t >> 5) & 15;
  int ks   = t >> 9;
  int n    = lane & 15;
  int kb   = (lane >> 4) * 16;
  const float* src = W + (size_t)(ft * 16 + n) * EMBED + ks * 32 + kb;
  v16bf hi, lo;
#pragma unroll
  for (int j = 0; j < 16; ++j) {
    float v  = src[j];
    bf16_t h = (bf16_t)v;            // RNE hi part
    float  r = v - (float)h;         // exact residual
    hi[j] = h;
    lo[j] = (bf16_t)r;
  }
  bf16_t* base = wpk + (size_t)ks * STEP_HALVES;
  *(v16bf*)(base + ((size_t)(0 * FTILES + ft) * 32 + lane) * 16) = hi;
  *(v16bf*)(base + ((size_t)(1 * FTILES + ft) * 32 + lane) * 16) = lo;
}

// Split one A chunk (16 fp32 per lane) into bf16 hi/lo fragments.
__device__ __forceinline__ void split_a(const v4f a0, const v4f a1,
                                        const v4f a2, const v4f a3,
                                        v16bf& hi, v16bf& lo) {
  float vals[16] = {a0.x, a0.y, a0.z, a0.w, a1.x, a1.y, a1.z, a1.w,
                    a2.x, a2.y, a2.z, a2.w, a3.x, a3.y, a3.z, a3.w};
#pragma unroll
  for (int j = 0; j < 16; ++j) {
    bf16_t h = (bf16_t)vals[j];
    hi[j] = h;
    lo[j] = (bf16_t)(vals[j] - (float)h);
  }
}

// ---------------------------------------------------------------------------
// Kernel 2: 512 threads = 16 waves = 2 M-groups x 8 F-groups over 128 rows.
// Wave(mg,fg): M-tiles mg*4..mg*4+3, f-tiles fg*2..fg*2+1.
// Waves 0..7 additionally convert+stage A tile wv each K-step.
// ---------------------------------------------------------------------------
__global__ __launch_bounds__(512, 1) void inv_patch_gemm_kernel(
    const float* __restrict__ x, const bf16_t* __restrict__ wpk,
    const float* __restrict__ bias, float* __restrict__ out) {
  __shared__ __align__(16) bf16_t Bb[STEP_HALVES];       // 32KB
  __shared__ __align__(16) bf16_t Ab[ATILES][2][32 * 16]; // 16KB

  const int tid  = threadIdx.x;
  const int lane = tid & 31;
  const int wv   = tid >> 5;          // 0..15
  const int mg   = wv & (MGROUPS - 1);
  const int fg   = wv >> 1;           // 0..7
  const int rowbase = blockIdx.x * WGROWS;

  const uint4* wsrc = (const uint4*)wpk;

  // A-filler addressing (16x32 A tile, contiguous k order): lanes 0-15 -> M=lane,
  // K runs {0..15}; lanes 16-31 -> M=lane-16, K runs {16..31}.
  const int mrowf = rowbase + wv * 16 + (lane & 15);   // filler's row (wv<8)
  const int kbA   = (lane >> 4) * 16;
  const float* xrow = x + (size_t)mrowf * EMBED + kbA;

  v8f acc[MT][FT] = {};

  // ---- prologue: stage A(0) and B(0), then barrier ----
  if (wv < ATILES) {
    v4f a0 = *(const v4f*)(xrow);
    v4f a1 = *(const v4f*)(xrow + 4);
    v4f a2 = *(const v4f*)(xrow + 8);
    v4f a3 = *(const v4f*)(xrow + 12);
    v16bf hi, lo;
    split_a(a0, a1, a2, a3, hi, lo);
    *(v16bf*)(&Ab[wv][0][lane * 16]) = hi;
    *(v16bf*)(&Ab[wv][1][lane * 16]) = lo;
  }
  {
    uint4 st[4];
#pragma unroll
    for (int j = 0; j < 4; ++j) st[j] = wsrc[tid * 4 + j];
    uint4* dst = (uint4*)(&Bb[0]);
#pragma unroll
    for (int j = 0; j < 4; ++j) dst[tid * 4 + j] = st[j];
  }
  __syncthreads();

  for (int ks = 0; ks < KSTEPS; ++ks) {
    // Warm WGP$ for next step's B staging (global_prefetch_b8).
    if (ks + 1 < KSTEPS)
      __builtin_prefetch(wsrc + (size_t)(ks + 1) * STEP_UINT4 + tid * 4, 0, 3);

    // Filler waves: issue next A-chunk loads early (ready by the refill phase).
    v4f gA0, gA1, gA2, gA3;
    if (wv < ATILES && ks + 1 < KSTEPS) {
      const float* pn = xrow + (ks + 1) * 32;
      gA0 = *(const v4f*)(pn);
      gA1 = *(const v4f*)(pn + 4);
      gA2 = *(const v4f*)(pn + 8);
      gA3 = *(const v4f*)(pn + 12);
    }

    // Load all 4 A tiles' hi/lo fragments into registers (resident this step).
    v16bf ahi[MT], alo[MT];
#pragma unroll
    for (int tt = 0; tt < MT; ++tt) {
      const bf16_t* ab = &Ab[mg * MT + tt][0][0];
      ahi[tt] = *(const v16bf*)(ab + lane * 16);
      alo[tt] = *(const v16bf*)(ab + 512 + lane * 16);
    }

    // Stream B: each fragment pair feeds 12 WMMAs (4 M-tiles x 3 terms).
#pragma unroll
    for (int fl = 0; fl < FT; ++fl) {
      const int ft = fg * FT + fl;
      v16bf bhi = *(const v16bf*)(&Bb[((size_t)ft * 32 + lane) * 16]);
      v16bf blo = *(const v16bf*)(&Bb[((size_t)(FTILES + ft) * 32 + lane) * 16]);
#pragma unroll
      for (int tt = 0; tt < MT; ++tt) {
        acc[tt][fl] = __builtin_amdgcn_wmma_f32_16x16x32_bf16(
            false, ahi[tt], false, bhi, (short)0, acc[tt][fl], false, false);
        acc[tt][fl] = __builtin_amdgcn_wmma_f32_16x16x32_bf16(
            false, alo[tt], false, bhi, (short)0, acc[tt][fl], false, false);
        acc[tt][fl] = __builtin_amdgcn_wmma_f32_16x16x32_bf16(
            false, ahi[tt], false, blo, (short)0, acc[tt][fl], false, false);
      }
    }

    // Refill A and B for K-step ks+1 (single-buffered: read -> bar -> write -> bar).
    if (ks + 1 < KSTEPS) {
      uint4 sn[4];
      const uint4* s2 = wsrc + (size_t)(ks + 1) * STEP_UINT4 + tid * 4;
#pragma unroll
      for (int j = 0; j < 4; ++j) sn[j] = s2[j];   // latency hides behind barrier
      __syncthreads();   // all reads of Ab/Bb for step ks complete
      uint4* dst = (uint4*)(&Bb[0]);
#pragma unroll
      for (int j = 0; j < 4; ++j) dst[tid * 4 + j] = sn[j];
      if (wv < ATILES) {
        v16bf hi, lo;
        split_a(gA0, gA1, gA2, gA3, hi, lo);
        *(v16bf*)(&Ab[wv][0][lane * 16]) = hi;
        *(v16bf*)(&Ab[wv][1][lane * 16]) = lo;
      }
      __syncthreads();   // publish before compute(ks+1)
    }
  }

  // ---- epilogue: bias + voxel-shuffle scatter (store-once, linear index) ----
  // D layout: lanes 0-15 -> M=r, N=lane; lanes 16-31 -> M=8+r, N=lane-16.
  const int ncol    = lane & 15;
  const int halfsel = lane >> 4;
#pragma unroll
  for (int tt = 0; tt < MT; ++tt) {
    const int mbase = rowbase + (mg * MT + tt) * 16;
    int rbase[8];
#pragma unroll
    for (int r = 0; r < 8; ++r) {
      int m    = mbase + halfsel * 8 + r;   // global row in [0, 65536)
      int bidx = m >> 15;                   // batch
      int nidx = m & 32767;                 // voxel index
      int i    = nidx >> 10;
      int j    = (nidx >> 5) & 31;
      int kk   = nidx & 31;
      rbase[r] = ((bidx * 512 + i * 4) * 128 + j * 4) * 128 + kk * 4;
    }
#pragma unroll
    for (int fl = 0; fl < FT; ++fl) {
      int   f  = (fg * FT + fl) * 16 + ncol;
      float bv = bias[f];
      int c = f >> 6, a = (f >> 4) & 3, d = (f >> 2) & 3, e = f & 3;
      int foff = ((c * 128 + a) * 128 + d) * 128 + e;
#pragma unroll
      for (int r = 0; r < 8; ++r) {
        out[(size_t)(rbase[r] + foff)] = acc[tt][fl][r] + bv;
      }
    }
  }
}

// ---------------------------------------------------------------------------
extern "C" void kernel_launch(void* const* d_in, const int* in_sizes, int n_in,
                              void* d_out, int out_size, void* d_ws, size_t ws_size,
                              hipStream_t stream) {
  const float* x = (const float*)d_in[0];
  const float* W = (const float*)d_in[1];
  const float* b = (const float*)d_in[2];
  float* out = (float*)d_out;

  bf16_t* wpk = (bf16_t*)d_ws;   // 24 * 32KB = 768KB packed W fragments

  const int packThreads = KSTEPS * FTILES * 32;   // 12288
  pack_w_kernel<<<packThreads / 256, 256, 0, stream>>>(W, wpk);

  const int Mrows = in_sizes[0] / EMBED;          // 65536
  inv_patch_gemm_kernel<<<Mrows / WGROWS, 512, 0, stream>>>(x, wpk, b, out);
}
